// VisionMambaNet_66460323938991
// MI455X (gfx1250) — matrix operand order repaired and checked by reading, not compiled, not run on hardware
//
#include <hip/hip_runtime.h>
#include <math.h>

// ---------------------------------------------------------------- constants
constexpr int EMBED    = 384;
constexpr int D_INNER  = 768;
constexpr int D_STATE  = 16;
constexpr int DT_RANK  = 24;
constexpr int DEPTH    = 24;
constexpr int IMG      = 512;
constexpr int PATCH    = 16;
constexpr int STRIDE_P = 8;
constexpr int GRID_P   = (IMG - PATCH) / STRIDE_P + 1;   // 63
constexpr int NPATCH   = GRID_P * GRID_P;                // 3969
constexpr int TOKEN_POS= NPATCH / 2;                     // 1984
constexpr int SEQ      = NPATCH + 1;                     // 3970

// ---------------------------------------------------------------- types
typedef float          v8f   __attribute__((ext_vector_type(8)));
typedef __bf16         v16bf __attribute__((ext_vector_type(16)));
typedef unsigned short v8us  __attribute__((ext_vector_type(8)));
typedef unsigned short v16us __attribute__((ext_vector_type(16)));
typedef int            v4i   __attribute__((ext_vector_type(4)));

#define BM 128
#define BN 128
#define KB 32

// gfx1250 async global->LDS DMA path (ASYNCcnt), guarded so either toolchain compiles
#if defined(__HIP_DEVICE_COMPILE__) && defined(__gfx1250__)
#if __has_builtin(__builtin_amdgcn_global_load_async_to_lds_b128)
#define HAVE_ASYNC_LDS 1
#endif
#endif

__device__ __forceinline__ void wait_async0() {
#if defined(HAVE_ASYNC_LDS)
#if __has_builtin(__builtin_amdgcn_s_wait_asynccnt)
  __builtin_amdgcn_s_wait_asynccnt(0);
#else
  asm volatile("s_wait_asynccnt 0x0" ::: "memory");
#endif
#endif
}

__device__ __forceinline__ unsigned short f32_to_bf16_rne(float f) {
  union { float f; unsigned int u; } x; x.f = f;
  unsigned int u = x.u;
  unsigned int r = u + 0x7FFFu + ((u >> 16) & 1u);
  return (unsigned short)(r >> 16);
}
__device__ __forceinline__ float bf16_to_f32(unsigned short u) {
  union { unsigned int u; float f; } x; x.u = ((unsigned int)u) << 16;
  return x.f;
}

// ================================================================ f32 -> bf16 blob convert
__global__ void cvt_bf16_kernel(const float* __restrict__ src, unsigned short* __restrict__ dst,
                                long long n)
{
  long long i = (long long)blockIdx.x * blockDim.x + threadIdx.x;
  if (i < n) dst[i] = f32_to_bf16_rne(src[i]);
}

// ================================================================ GEMM (bf16 in, f32 accum)
// C[M,N] = act( A[M,K] * W[N,K]^T + bias ) * alpha ; optional bf16 mirror output Cbf
// act: 0 = none, 1 = softplus
__global__ __launch_bounds__(256)
void gemm_bf16_wmma(const unsigned short* __restrict__ A, const unsigned short* __restrict__ W,
                    float* __restrict__ C, unsigned short* __restrict__ Cbf,
                    int M, int N, int K, int lda, int ldw, int ldc, int ldcbf,
                    const float* __restrict__ bias, int act, float alpha)
{
  __shared__ __align__(16) unsigned short As[BM * KB];
  __shared__ __align__(16) unsigned short Bs[BN * KB];

  const int tid   = threadIdx.x;
  const int lane  = tid & 31;
  const int wave  = tid >> 5;        // 0..7
  const int wm    = wave >> 1;       // 0..3  -> 32 rows each
  const int wn    = wave & 1;        // 0..1  -> 64 cols each
  const int mlan  = lane & 15;
  const int lh    = lane >> 4;
  const int rowBase = blockIdx.y * BM;
  const int colBase = blockIdx.x * BN;

  v8f acc[2][4] = {};

  for (int k0 = 0; k0 < K; k0 += KB) {
    // ---- stage A tile (BM x KB bf16): 512 16-byte chunks, 2 per thread
#pragma unroll
    for (int it = 0; it < 2; ++it) {
      int chunk = tid + it * 256;
      int r  = chunk >> 2;                 // 0..127
      int kc = (chunk & 3) << 3;           // 0,8,16,24 (bf16 elems)
      bool ok = (rowBase + r) < M && (k0 + kc) < K;
#ifdef HAVE_ASYNC_LDS
      if (ok) {
        __builtin_amdgcn_global_load_async_to_lds_b128(
            (__attribute__((address_space(1))) v4i*)(A + (size_t)(rowBase + r) * lda + k0 + kc),
            (__attribute__((address_space(3))) v4i*)&As[r * KB + kc], 0, 0);
      } else {
        v8us z = {};
        *(v8us*)&As[r * KB + kc] = z;
      }
#else
      v8us v = {};
      if (ok) v = *(const v8us*)(A + (size_t)(rowBase + r) * lda + k0 + kc);
      *(v8us*)&As[r * KB + kc] = v;
#endif
    }
    // ---- stage W tile (BN x KB bf16)
#pragma unroll
    for (int it = 0; it < 2; ++it) {
      int chunk = tid + it * 256;
      int r  = chunk >> 2;
      int kc = (chunk & 3) << 3;
      bool ok = (colBase + r) < N && (k0 + kc) < K;
#ifdef HAVE_ASYNC_LDS
      if (ok) {
        __builtin_amdgcn_global_load_async_to_lds_b128(
            (__attribute__((address_space(1))) v4i*)(W + (size_t)(colBase + r) * ldw + k0 + kc),
            (__attribute__((address_space(3))) v4i*)&Bs[r * KB + kc], 0, 0);
      } else {
        v8us z = {};
        *(v8us*)&Bs[r * KB + kc] = z;
      }
#else
      v8us v = {};
      if (ok) v = *(const v8us*)(W + (size_t)(colBase + r) * ldw + k0 + kc);
      *(v8us*)&Bs[r * KB + kc] = v;
#endif
    }
    wait_async0();
    __syncthreads();

    // ---- fragments per CDNA5 WMMA VGPR layouts
    v16bf afrag[2];
#pragma unroll
    for (int tm = 0; tm < 2; ++tm) {
      int r = wm * 32 + tm * 16 + mlan;
      v8us lo = *(const v8us*)&As[r * KB + lh * 8];        // K = lh*8 .. +7
      v8us hi = *(const v8us*)&As[r * KB + lh * 8 + 16];   // K = lh*8+16 .. +7
      v16us a = __builtin_shufflevector(lo, hi, 0,1,2,3,4,5,6,7,8,9,10,11,12,13,14,15);
      afrag[tm] = __builtin_bit_cast(v16bf, a);
    }
    v16bf bfrag[4];
#pragma unroll
    for (int tn = 0; tn < 4; ++tn) {
      int r = wn * 64 + tn * 16 + mlan;
      v8us lo = *(const v8us*)&Bs[r * KB + lh * 8];        // K = lh*8 .. +7
      v8us hi = *(const v8us*)&Bs[r * KB + lh * 8 + 16];   // K = lh*8+16 .. +7
      v16us b = __builtin_shufflevector(lo, hi, 0,1,2,3,4,5,6,7,8,9,10,11,12,13,14,15);
      bfrag[tn] = __builtin_bit_cast(v16bf, b);
    }
#pragma unroll
    for (int tm = 0; tm < 2; ++tm)
#pragma unroll
      for (int tn = 0; tn < 4; ++tn)
        acc[tm][tn] = __builtin_amdgcn_wmma_f32_16x16x32_bf16(
            false, afrag[tm], false, bfrag[tn], (short)0, acc[tm][tn], false, false);
    __syncthreads();
  }

  // ---- epilogue (C/D layout: VGPR r -> M = r + 8*lh, N = lane&15)
#pragma unroll
  for (int tm = 0; tm < 2; ++tm) {
#pragma unroll
    for (int tn = 0; tn < 4; ++tn) {
      int n = colBase + wn * 64 + tn * 16 + mlan;
#pragma unroll
      for (int r = 0; r < 8; ++r) {
        int m = rowBase + wm * 32 + tm * 16 + r + lh * 8;
        if (m < M && n < N) {
          float v = acc[tm][tn][r];
          if (bias) v += bias[n];
          if (act == 1) v = (v > 20.f) ? v : log1pf(__expf(v));  // softplus
          v *= alpha;
          if (C)   C[(size_t)m * ldc + n] = v;
          if (Cbf) Cbf[(size_t)m * ldcbf + n] = f32_to_bf16_rne(v);
        }
      }
    }
  }
}

// ================================================================ im2col (writes bf16)
__global__ void im2col_kernel(const float* __restrict__ x, unsigned short* __restrict__ dst)
{
  int idx = blockIdx.x * blockDim.x + threadIdx.x;
  if (idx >= NPATCH * 768) return;
  int p   = idx / 768;
  int col = idx - p * 768;
  int c   = col >> 8;
  int rem = col & 255;
  int py  = rem >> 4, px = rem & 15;
  int gy  = p / GRID_P, gx = p - gy * GRID_P;
  float v = x[(size_t)c * IMG * IMG + (size_t)(gy * STRIDE_P + py) * IMG + gx * STRIDE_P + px];
  dst[idx] = f32_to_bf16_rne(v);
}

// ================================================================ token assembly
__global__ void assemble_kernel(const float* __restrict__ tok, const float* __restrict__ cls,
                                const float* __restrict__ pos, float* __restrict__ h,
                                float* __restrict__ res)
{
  int idx = blockIdx.x * blockDim.x + threadIdx.x;
  if (idx >= SEQ * EMBED) return;
  int l = idx / EMBED;
  int e = idx - l * EMBED;
  float t;
  if (l == TOKEN_POS) t = cls[e];
  else {
    int p = (l < TOKEN_POS) ? l : (l - 1);
    t = tok[(size_t)p * EMBED + e];
  }
  h[idx]   = t + pos[idx];
  res[idx] = 0.f;
}

// ================================================================ res += h; hn = rmsnorm(res)*w
__global__ __launch_bounds__(128)
void addnorm_kernel(const float* __restrict__ h, float* __restrict__ res,
                    float* __restrict__ hn, unsigned short* __restrict__ hnb,
                    const float* __restrict__ w)
{
  __shared__ float wsum[4];
  int row = blockIdx.x;
  int tid = threadIdx.x;
  int lane = tid & 31, wave = tid >> 5;
  float v[3]; float s = 0.f;
#pragma unroll
  for (int jj = 0; jj < 3; ++jj) {
    int j = tid + jj * 128;
    float t = h[(size_t)row * EMBED + j] + res[(size_t)row * EMBED + j];
    res[(size_t)row * EMBED + j] = t;
    v[jj] = t; s += t * t;
  }
#pragma unroll
  for (int m = 16; m > 0; m >>= 1) s += __shfl_xor(s, m, 32);
  if (lane == 0) wsum[wave] = s;
  __syncthreads();
  float total = wsum[0] + wsum[1] + wsum[2] + wsum[3];
  float scale = rsqrtf(total / (float)EMBED + 1e-5f);
#pragma unroll
  for (int jj = 0; jj < 3; ++jj) {
    int j = tid + jj * 128;
    float o = v[jj] * scale * w[j];
    hn[(size_t)row * EMBED + j]  = o;
    hnb[(size_t)row * EMBED + j] = f32_to_bf16_rne(o);
  }
}

// ================================================================ causal conv1d + SiLU -> bf16
__global__ void conv_silu_kernel(const float* __restrict__ xz, const float* __restrict__ w,
                                 const float* __restrict__ b, unsigned short* __restrict__ xc,
                                 int rev)
{
  int idx = blockIdx.x * blockDim.x + threadIdx.x;
  if (idx >= SEQ * D_INNER) return;
  int l = idx / D_INNER;
  int d = idx - l * D_INNER;
  float acc = b[d];
#pragma unroll
  for (int k = 0; k < 4; ++k) {
    int ls = l - 3 + k;
    if (ls >= 0) {
      int lg = rev ? (SEQ - 1 - ls) : ls;
      acc += xz[(size_t)lg * 1536 + d] * w[d * 4 + k];
    }
  }
  acc = acc / (1.f + __expf(-acc));   // silu
  xc[idx] = f32_to_bf16_rne(acc);
}

// ================================================================ selective scan (+ gate + write)
// Thread owns state (d, s). 16-lane shfl reduction over s (wave32 halves).
// mode accumulate=0: write f32 ysum ; accumulate=1: ysum_bf = bf16(ysum + y)
__global__ __launch_bounds__(256)
void scan_kernel(const unsigned short* __restrict__ xc, const float* __restrict__ delta,
                 const unsigned short* __restrict__ xdbl, const float* __restrict__ xz,
                 const float* __restrict__ A_log, const float* __restrict__ Dv,
                 float* __restrict__ ysum, unsigned short* __restrict__ ysum_bf,
                 int rev, int accumulate)
{
  int lane = threadIdx.x & 31;
  int wave = threadIdx.x >> 5;
  int s    = lane & 15;
  int d    = blockIdx.x * 16 + wave * 2 + (lane >> 4);
  float Aval = -__expf(A_log[(size_t)d * D_STATE + s]);
  float Dd   = Dv[d];
  float hst  = 0.f;
  for (int l = 0; l < SEQ; ++l) {
    float dlt = delta[(size_t)l * D_INNER + d];
    float xv  = bf16_to_f32(xc[(size_t)l * D_INNER + d]);
    float Bv  = bf16_to_f32(xdbl[(size_t)l * 56 + DT_RANK + s]);
    float Cv  = bf16_to_f32(xdbl[(size_t)l * 56 + DT_RANK + D_STATE + s]);
    float dA  = __expf(dlt * Aval);
    hst = hst * dA + dlt * xv * Bv;
    float c = hst * Cv;
    c += __shfl_xor(c, 1, 32);
    c += __shfl_xor(c, 2, 32);
    c += __shfl_xor(c, 4, 32);
    c += __shfl_xor(c, 8, 32);
    if (s == 0) {
      int lg = rev ? (SEQ - 1 - l) : l;
      float z = xz[(size_t)lg * 1536 + D_INNER + d];
      float y = (c + xv * Dd) * (z / (1.f + __expf(-z)));
      size_t o = (size_t)lg * D_INNER + d;
      if (accumulate) ysum_bf[o] = f32_to_bf16_rne(ysum[o] + y);
      else            ysum[o] = y;
    }
  }
}

// ================================================================ classifier head
__global__ __launch_bounds__(256)
void head_kernel(const float* __restrict__ hf, const float* __restrict__ W1,
                 const float* __restrict__ b1, const float* __restrict__ W2,
                 const float* __restrict__ b2, float* __restrict__ out)
{
  __shared__ float hd[256];
  const float* c = hf + (size_t)TOKEN_POS * EMBED;
  int j = threadIdx.x;
  float a = b1[j];
  for (int k = 0; k < EMBED; ++k) a += c[k] * W1[j * EMBED + k];
  hd[j] = a > 0.f ? a : 0.f;
  __syncthreads();
  if (j < 10) {
    float o = b2[j];
    for (int k = 0; k < 256; ++k) o += hd[k] * W2[j * 256 + k];
    out[j] = o;
  }
}

// ================================================================ host side
static inline void launch_gemm(hipStream_t s, const unsigned short* A, int lda,
                               const unsigned short* W, float* C, int ldc,
                               unsigned short* Cbf, int ldcbf,
                               int M, int N, int K, const float* bias, int act, float alpha)
{
  dim3 grid((N + BN - 1) / BN, (M + BM - 1) / BM);
  gemm_bf16_wmma<<<grid, 256, 0, s>>>(A, W, C, Cbf, M, N, K, lda, K, ldc, ldcbf, bias, act, alpha);
}

extern "C" void kernel_launch(void* const* d_in, const int* in_sizes, int n_in,
                              void* d_out, int out_size, void* d_ws, size_t ws_size,
                              hipStream_t stream)
{
  (void)in_sizes; (void)n_in; (void)out_size; (void)ws_size;
  const float* x           = (const float*)d_in[0];
  const float* patch_w     = (const float*)d_in[1];
  const float* patch_b     = (const float*)d_in[2];
  const float* cls_tok     = (const float*)d_in[3];
  const float* pos_emb     = (const float*)d_in[4];
  const float* in_proj_w   = (const float*)d_in[5];
  const float* conv_w      = (const float*)d_in[6];
  const float* conv_b      = (const float*)d_in[7];
  const float* x_proj_w    = (const float*)d_in[8];
  const float* dt_proj_w   = (const float*)d_in[9];
  const float* dt_proj_b   = (const float*)d_in[10];
  const float* A_log       = (const float*)d_in[11];
  const float* Dv          = (const float*)d_in[12];
  const float* conv_bw     = (const float*)d_in[13];
  const float* conv_bb     = (const float*)d_in[14];
  const float* x_proj_bw   = (const float*)d_in[15];
  const float* dt_proj_bw  = (const float*)d_in[16];
  const float* dt_proj_bb  = (const float*)d_in[17];
  const float* A_b_log     = (const float*)d_in[18];
  const float* D_b         = (const float*)d_in[19];
  const float* out_proj_w  = (const float*)d_in[20];
  const float* norm_w      = (const float*)d_in[21];
  const float* norm_f_w    = (const float*)d_in[22];
  const float* head_w1     = (const float*)d_in[23];
  const float* head_b1     = (const float*)d_in[24];
  const float* head_w2     = (const float*)d_in[25];
  const float* head_b2     = (const float*)d_in[26];

  char* base = (char*)d_ws;
  size_t off = 0;
  auto carve = [&](size_t bytes) -> char* {
    char* p = base + off;
    off = (off + bytes + 255) & ~(size_t)255;
    return p;
  };
  // activations
  unsigned short* im2b  = (unsigned short*)carve((size_t)NPATCH * 768 * 2);
  float*          tok   = (float*)carve((size_t)NPATCH * EMBED * 4);
  float*          h     = (float*)carve((size_t)SEQ * EMBED * 4);
  float*          res   = (float*)carve((size_t)SEQ * EMBED * 4);
  float*          hn    = (float*)carve((size_t)SEQ * EMBED * 4);
  unsigned short* hnb   = (unsigned short*)carve((size_t)SEQ * EMBED * 2);
  float*          xz    = (float*)carve((size_t)SEQ * 2 * D_INNER * 4);
  unsigned short* xcb   = (unsigned short*)carve((size_t)SEQ * D_INNER * 2);
  unsigned short* xdblb = (unsigned short*)carve((size_t)SEQ * 56 * 2);
  float*          dlt   = (float*)carve((size_t)SEQ * D_INNER * 4);
  float*          ysum  = (float*)carve((size_t)SEQ * D_INNER * 4);
  unsigned short* ysumb = (unsigned short*)carve((size_t)SEQ * D_INNER * 2);
  // bf16 weight blobs (converted once per launch; resident in 192MB L2 afterwards)
  const size_t n_patchw = (size_t)EMBED * 768;
  const size_t n_inproj = (size_t)DEPTH * 2 * D_INNER * EMBED;
  const size_t n_xproj  = (size_t)DEPTH * 56 * D_INNER;
  const size_t n_dtproj = (size_t)DEPTH * D_INNER * DT_RANK;
  const size_t n_outprj = (size_t)DEPTH * EMBED * D_INNER;
  unsigned short* wpatch = (unsigned short*)carve(n_patchw * 2);
  unsigned short* win    = (unsigned short*)carve(n_inproj * 2);
  unsigned short* wxp    = (unsigned short*)carve(n_xproj * 2);
  unsigned short* wxpb   = (unsigned short*)carve(n_xproj * 2);
  unsigned short* wdt    = (unsigned short*)carve(n_dtproj * 2);
  unsigned short* wdtb   = (unsigned short*)carve(n_dtproj * 2);
  unsigned short* wout   = (unsigned short*)carve(n_outprj * 2);

  auto cvt = [&](const float* s, unsigned short* d, size_t n) {
    cvt_bf16_kernel<<<(unsigned)((n + 255) / 256), 256, 0, stream>>>(s, d, (long long)n);
  };
  cvt(patch_w, wpatch, n_patchw);
  cvt(in_proj_w, win, n_inproj);
  cvt(x_proj_w, wxp, n_xproj);
  cvt(x_proj_bw, wxpb, n_xproj);
  cvt(dt_proj_w, wdt, n_dtproj);
  cvt(dt_proj_bw, wdtb, n_dtproj);
  cvt(out_proj_w, wout, n_outprj);

  // ---- patch embedding: im2col + WMMA GEMM + token assembly
  im2col_kernel<<<(NPATCH * 768 + 255) / 256, 256, 0, stream>>>(x, im2b);
  launch_gemm(stream, im2b, 768, wpatch, tok, EMBED, nullptr, 0,
              NPATCH, EMBED, 768, patch_b, 0, 1.f);
  assemble_kernel<<<(SEQ * EMBED + 255) / 256, 256, 0, stream>>>(tok, cls_tok, pos_emb, h, res);

  // ---- 24 bidirectional Mamba blocks
  for (int l = 0; l < DEPTH; ++l) {
    addnorm_kernel<<<SEQ, 128, 0, stream>>>(h, res, hn, hnb, norm_w + (size_t)l * EMBED);
    launch_gemm(stream, hnb, EMBED, win + (size_t)l * 2 * D_INNER * EMBED,
                xz, 2 * D_INNER, nullptr, 0, SEQ, 2 * D_INNER, EMBED, nullptr, 0, 1.f);

    // forward direction
    conv_silu_kernel<<<(SEQ * D_INNER + 255) / 256, 256, 0, stream>>>(
        xz, conv_w + (size_t)l * D_INNER * 4, conv_b + (size_t)l * D_INNER, xcb, 0);
    launch_gemm(stream, xcb, D_INNER, wxp + (size_t)l * 56 * D_INNER,
                nullptr, 0, xdblb, 56, SEQ, 56, D_INNER, nullptr, 0, 1.f);
    launch_gemm(stream, xdblb, 56, wdt + (size_t)l * D_INNER * DT_RANK,
                dlt, D_INNER, nullptr, 0, SEQ, D_INNER, DT_RANK,
                dt_proj_b + (size_t)l * D_INNER, 1, 1.f);
    scan_kernel<<<D_INNER / 16, 256, 0, stream>>>(
        xcb, dlt, xdblb, xz, A_log + (size_t)l * D_INNER * D_STATE,
        Dv + (size_t)l * D_INNER, ysum, ysumb, 0, 0);

    // backward direction (reversed coordinates)
    conv_silu_kernel<<<(SEQ * D_INNER + 255) / 256, 256, 0, stream>>>(
        xz, conv_bw + (size_t)l * D_INNER * 4, conv_bb + (size_t)l * D_INNER, xcb, 1);
    launch_gemm(stream, xcb, D_INNER, wxpb + (size_t)l * 56 * D_INNER,
                nullptr, 0, xdblb, 56, SEQ, 56, D_INNER, nullptr, 0, 1.f);
    launch_gemm(stream, xdblb, 56, wdtb + (size_t)l * D_INNER * DT_RANK,
                dlt, D_INNER, nullptr, 0, SEQ, D_INNER, DT_RANK,
                dt_proj_bb + (size_t)l * D_INNER, 1, 1.f);
    scan_kernel<<<D_INNER / 16, 256, 0, stream>>>(
        xcb, dlt, xdblb, xz, A_b_log + (size_t)l * D_INNER * D_STATE,
        D_b + (size_t)l * D_INNER, ysum, ysumb, 1, 1);

    // out projection, /2, becomes new h
    launch_gemm(stream, ysumb, D_INNER, wout + (size_t)l * EMBED * D_INNER,
                h, EMBED, nullptr, 0, SEQ, EMBED, D_INNER, nullptr, 0, 0.5f);
  }

  // ---- final norm + head
  addnorm_kernel<<<SEQ, 128, 0, stream>>>(h, res, hn, hnb, norm_f_w);
  head_kernel<<<1, 256, 0, stream>>>(hn, head_w1, head_b1, head_w2, head_b2, (float*)d_out);
}
